// CharLSTMLanguageModelPack_43937515438304
// MI455X (gfx1250) — matrix operand order repaired and checked by reading, not compiled, not run on hardware
//
#include <hip/hip_runtime.h>

// CDNA5 / gfx1250 — rhyme soft-DTW loss.
// One wave32 per batch: softmax (2-pass, cached reload) -> f16 A fragments,
// gathered f16 B fragments from phon_cost -> 8x v_wmma_f32_16x16x32_f16,
// LDS-staged 16x16 tile -> anti-diagonal systolic softmin DP via shuffles.

typedef __attribute__((ext_vector_type(16))) _Float16 v16h;
typedef __attribute__((ext_vector_type(8)))  float    v8f;

#define TDIM 16
#define MDIM 16
#define VDIM 256
#define INS_DEL_C 10.0f
#define FIRST_CHAR_COST_C 10.0f

__global__ __launch_bounds__(256)
void rhyme_dp_wmma_kernel(const float* __restrict__ logits,
                          const int*   __restrict__ tidx,
                          const float* __restrict__ phon,
                          float*       __restrict__ out,
                          int Btot)
{
  // one 16x16 sub tile per wave (8 waves / block), padded to dodge bank conflicts
  __shared__ float subs[8][16][17];

  const int lane = threadIdx.x & 31;
  const int w    = threadIdx.x >> 5;
  const int b    = blockIdx.x * 8 + w;
  if (b >= Btot) return;                 // wave-uniform: EXEC stays all-ones

  const int t = lane & 15;               // row of A / column of B & DP
  const int s = lane >> 4;               // half-select (K sub-block / row-half)
  const float* Lrow = logits + ((size_t)b * TDIM + t) * VDIM;

  // ---------- pass 1: row max over this lane's 128 columns, merge halves ----------
  float mx = -3.402823466e38f;
  #pragma unroll
  for (int g = 0; g < 16; ++g) {
    const float4* p4 = (const float4*)(Lrow + g * 16 + s * 8);
    float4 x0 = p4[0], x1 = p4[1];
    mx = fmaxf(mx, fmaxf(fmaxf(x0.x, x0.y), fmaxf(x0.z, x0.w)));
    mx = fmaxf(mx, fmaxf(fmaxf(x1.x, x1.y), fmaxf(x1.z, x1.w)));
  }
  mx = fmaxf(mx, __shfl_xor(mx, 16, 32));  // full-row max on both half-lanes

  // ---------- pass 2: exp(x-max) straight into f16 A fragments, accumulate sum ----------
  // A (16x32 f16) layout: lane group s covers K%16 in [8s, 8s+8); fragment element
  // (g&1)*8 + j  <->  K_local = (g&1)*16 + 8*s + j  of fragment kb = g>>1.
  v16h afrag[8];
  float sum = 0.0f;
  #pragma unroll
  for (int g = 0; g < 16; ++g) {
    const float4* p4 = (const float4*)(Lrow + g * 16 + s * 8);
    float4 x0 = p4[0], x1 = p4[1];
    float e0 = __expf(x0.x - mx), e1 = __expf(x0.y - mx);
    float e2 = __expf(x0.z - mx), e3 = __expf(x0.w - mx);
    float e4 = __expf(x1.x - mx), e5 = __expf(x1.y - mx);
    float e6 = __expf(x1.z - mx), e7 = __expf(x1.w - mx);
    sum += ((e0 + e1) + (e2 + e3)) + ((e4 + e5) + (e6 + e7));
    const int k = g >> 1, h = (g & 1) * 8;
    afrag[k][h + 0] = (_Float16)e0;  afrag[k][h + 1] = (_Float16)e1;
    afrag[k][h + 2] = (_Float16)e2;  afrag[k][h + 3] = (_Float16)e3;
    afrag[k][h + 4] = (_Float16)e4;  afrag[k][h + 5] = (_Float16)e5;
    afrag[k][h + 6] = (_Float16)e6;  afrag[k][h + 7] = (_Float16)e7;
  }
  sum += __shfl_xor(sum, 16, 32);          // full-row sum on both half-lanes
  const float    inv  = 1.0f / sum;
  const _Float16 hinv = (_Float16)inv;
  #pragma unroll
  for (int k = 0; k < 8; ++k) {
    #pragma unroll
    for (int e = 0; e < 16; ++e) afrag[k][e] *= hinv;   // -> v_pk_mul_f16 pairs
  }

  // ---------- B gather + 8x WMMA:  sub = p @ phon[:, idx]  (16x16 f32) ----------
  // B (32x16 f16): lane group s holds K_local = (e>>3)*16 + 8*s + (e&7) for column m = lane&15.
  const int idxm = tidx[b * MDIM + t];
  v8f acc = {};
  #pragma unroll
  for (int kb = 0; kb < 8; ++kb) {
    v16h bf;
    #pragma unroll
    for (int e = 0; e < 16; ++e) {
      const int vr = kb * 32 + (e >> 3) * 16 + s * 8 + (e & 7);
      bf[e] = (_Float16)phon[vr * VDIM + idxm];          // L2-resident gather
    }
    acc = __builtin_amdgcn_wmma_f32_16x16x32_f16(
        /*neg_a=*/false, afrag[kb], /*neg_b=*/false, bf,
        /*c_mod=*/(short)0, acc, /*reuse_a=*/false, /*reuse_b=*/false);
  }

  // D layout: VGPR i = row (i + 8*s), lane = column (lane&15). Stage tile in LDS.
  #pragma unroll
  for (int i = 0; i < 8; ++i) subs[w][s * 8 + i][t] = acc[i];
  __syncthreads();

  // ---------- first-char probability term (row-0 stats live on lane 0) ----------
  const float m0 = __shfl(mx, 0, 32);
  const float s0 = __shfl(sum, 0, 32);
  const int   idx0 = tidx[b * MDIM + 0];
  const float p0 = __expf(logits[(size_t)b * TDIM * VDIM + idx0] - m0) / s0;

  // ---------- anti-diagonal systolic softmin DP over (T+1)x(M+1) ----------
  // Lane n (= lane&15) owns DP column j = n+1. At step s it computes cell
  // (i = s-n+1, j), pulling left/diag from lane n-1's last two values.
  float vcur = INS_DEL_C * (float)(t + 1);   // D[0][j] before activation
  float vold = 0.0f;
  #pragma unroll
  for (int step = 0; step < 31; ++step) {
    float l1 = __shfl_up(vcur, 1, 32);       // D[i][j-1]
    float l2 = __shfl_up(vold, 1, 32);       // D[i-1][j-1]
    if (t == 0) {                            // column-0 boundary
      l1 = INS_DEL_C * (float)(step + 1);
      l2 = INS_DEL_C * (float)step;
    }
    const unsigned r = (unsigned)(step - t);
    if (r < 16u) {
      const float sv = subs[w][r][t];
      const float aa = vcur + INS_DEL_C;     // up + ins
      const float bb = l1   + INS_DEL_C;     // left + del
      const float cc = l2   + sv;            // diag + sub
      const float mn = fminf(aa, fminf(bb, cc));
      const float v  = mn - __logf(__expf(mn - aa) + __expf(mn - bb) + __expf(mn - cc));
      vold = vcur;
      vcur = v;
    }
  }

  if (lane == 15)                            // D[16][16] finishes on lane 15
    out[b] = vcur + FIRST_CHAR_COST_C * (1.0f - p0);
}

extern "C" void kernel_launch(void* const* d_in, const int* in_sizes, int n_in,
                              void* d_out, int out_size, void* d_ws, size_t ws_size,
                              hipStream_t stream) {
  const float* logits = (const float*)d_in[0];   // (B, 16, 256) f32
  const int*   tidx   = (const int*)  d_in[1];   // (B, 16)      i32
  const float* phon   = (const float*)d_in[2];   // (256, 256)   f32
  float*       out    = (float*)d_out;           // (B,)         f32

  const int Btot   = in_sizes[0] / (TDIM * VDIM);
  const int blocks = (Btot + 7) / 8;             // 8 waves (batches) per block
  rhyme_dp_wmma_kernel<<<blocks, 256, 0, stream>>>(logits, tidx, phon, out, Btot);
}
